// LocalInfoProcess_41875931136257
// MI455X (gfx1250) — matrix operand order repaired and checked by reading, not compiled, not run on hardware
//
#include <hip/hip_runtime.h>

typedef _Float16 v16h __attribute__((ext_vector_type(16)));
typedef float    v8f  __attribute__((ext_vector_type(8)));

#define NIMG   512
#define NN     (512*512)
#define NBR    14
#define POOLW  170
#define POOLPER (POOLW*POOLW)            // 28900
#define OUTROW  (NBR + NBR*POOLPER)      // 404614
#define TEMP_BYTES ((size_t)NBR * NN * sizeof(float))

#define TILE_ROWS 16
#define TILE_COLS 128
#define HALO_R    21      // 16 + 5
#define HALO_C    133     // 128 + 5
#define LDS_STRIDE 136

__device__ __constant__ int d_KH[NBR] = {3,3,3,4,4,4,4,5,5,5,5,6,6,6};
__device__ __constant__ int d_KW[NBR] = {3,4,5,3,4,5,6,3,4,5,6,4,5,6};
static const int h_KH[NBR] = {3,3,3,4,4,4,4,5,5,5,5,6,6,6};
static const int h_KW[NBR] = {3,4,5,3,4,5,6,3,4,5,6,4,5,6};

struct KPtrs { const float* w[NBR]; };

__device__ __forceinline__ unsigned enc_f32(float f) {
    unsigned u = __float_as_uint(f);
    return (u & 0x80000000u) ? ~u : (u | 0x80000000u);
}

// ---------------------------------------------------------------------------
// Pack the 14 kernels into a 64x16 f16 B-matrix (K = tap in 6x6 patch, N = branch)
// ---------------------------------------------------------------------------
__global__ void lip_prep(KPtrs kp, _Float16* __restrict__ Bw) {
    int idx = blockIdx.x * 256 + threadIdx.x;
    if (idx >= 64 * 16) return;
    int k = idx >> 4, n = idx & 15;
    float v = 0.f;
    if (n < NBR && k < 36) {
        int pr = k / 6, pc = k % 6;
        int kh = d_KH[n], kw = d_KW[n];
        int ro = 2 - ((kh - 1) >> 1);        // embed offset of kernel in 6x6 patch
        int co = 2 - ((kw - 1) >> 1);
        int di = pr - ro, dj = pc - co;
        if (di >= 0 && di < kh && dj >= 0 && dj < kw)
            v = kp.w[n][di * kw + dj];
    }
    Bw[k * 16 + n] = (_Float16)v;
}

// ---------------------------------------------------------------------------
// temporal[br][i][j] = box-filter(|i-j|, kh x kw, SAME/zero-pad), closed form
// ---------------------------------------------------------------------------
__global__ void lip_temporal(float* __restrict__ t) {
    int idx = blockIdx.x * 256 + threadIdx.x;
    if (idx >= NBR * NN) return;
    int br  = idx >> 18;
    int rem = idx & (NN - 1);
    int i = rem >> 9, j = rem & 511;
    int kh = d_KH[br], kw = d_KW[br];
    int r0 = i - ((kh - 1) >> 1);
    int c0 = j - ((kw - 1) >> 1);
    int a  = max(c0, 0), b = min(c0 + kw - 1, NIMG - 1);
    int S = 0;
    if (a <= b) {
        int cnt = b - a + 1;
        for (int dr = 0; dr < kh; ++dr) {
            int r = r0 + dr;
            if (r < 0 || r >= NIMG) continue;
            if (b <= r)      S += r * cnt - (a + b) * cnt / 2;
            else if (a >= r) S += (a + b) * cnt / 2 - r * cnt;
            else {
                int l = r - a, rr = b - r;
                S += l * (l + 1) / 2 + rr * (rr + 1) / 2;
            }
        }
    }
    t[idx] = (float)S / (float)(kh * kw);
}

// ---------------------------------------------------------------------------
// Main fused kernel: BN -> 14 convs via WMMA (im2col over common 6x6 patch)
// -> *temporal -> global max (encoded atomicMax) + 3x3 pool (atomicAdd)
// grid: (512/128, 512/16, 16 batches), 256 threads = 8 waves
// ---------------------------------------------------------------------------
__global__ __launch_bounds__(256) void lip_main(
    const float* __restrict__ x, const float* __restrict__ biases,
    const float* __restrict__ gamma, const float* __restrict__ beta,
    const float* __restrict__ mmean, const float* __restrict__ mvar,
    const float* __restrict__ temporal, const _Float16* __restrict__ Bw,
    float* __restrict__ out)
{
    __shared__ _Float16 sx[HALO_R][LDS_STRIDE];
    __shared__ _Float16 sB[64 * 16];
    __shared__ float    smax[8][16];

    const int bidx = blockIdx.z;
    const int i0 = blockIdx.y * TILE_ROWS;
    const int j0 = blockIdx.x * TILE_COLS;
    const int tid = threadIdx.x;

    const float scale = gamma[0] * rsqrtf(mvar[0] + 1e-3f);
    const float shift = beta[0] - mmean[0] * scale;

    // stage BN-applied f16 tile with halo
    const float* xb = x + (size_t)bidx * NN;
    for (int idx = tid; idx < HALO_R * HALO_C; idx += 256) {
        int rr = idx / HALO_C, cc = idx % HALO_C;
        int r = i0 - 2 + rr, c = j0 - 2 + cc;
        float v = 0.f;
        if (r >= 0 && r < NIMG && c >= 0 && c < NIMG) v = xb[r * NIMG + c];
        sx[rr][cc] = (_Float16)(v * scale + shift);
    }
    for (int idx = tid; idx < 64 * 16; idx += 256) sB[idx] = Bw[idx];
    __syncthreads();

    const int lane = tid & 31;
    const int wv   = tid >> 5;
    const int g    = lane >> 4;       // lane group
    const int n    = lane & 15;       // branch column (D layout) / M pixel (A layout)
    const int jslice = j0 + wv * 16;  // this wave's 16-column slice

    // B fragments (invariant): lane L holds N=L%16, K = (h<8?h:h+8) + 8*(L/16) (+32 for chunk1)
    v16h bf0, bf1;
#pragma unroll
    for (int h = 0; h < 16; ++h) {
        bf0[h] = sB[((h < 8 ? h : h + 8) + 8 * g) * 16 + n];
        bf1[h] = sB[((h < 8 ? h : h + 8) + 8 * g + 32) * 16 + n];
    }

    const float bias = (n < NBR) ? biases[n] : 0.f;
    float lmax = -3.4e38f;

    for (int ir = 0; ir < TILE_ROWS; ++ir) {
        const int i = i0 + ir;
        // A fragments: lane holds pixel M = lane&15; K=(h<8?h:h+8)+8g (+32)
        const int mA = lane & 15;
        const int jl = wv * 16 + mA;      // lds col for pc=0
        v16h a0, a1;
#pragma unroll
        for (int h = 0; h < 16; ++h) {
            const int K0 = (h < 8 ? h : h + 8) + g * 8;   // 0..31, always a tap
            a0[h] = sx[ir + K0 / 6][jl + K0 % 6];
            const int K1 = K0 + 32;
            a1[h] = (K1 < 36) ? sx[ir + K1 / 6][jl + K1 % 6] : (_Float16)0.f;
        }
        v8f c = {0.f,0.f,0.f,0.f,0.f,0.f,0.f,0.f};
        c = __builtin_amdgcn_wmma_f32_16x16x32_f16(false, a0, false, bf0, (short)0, c, false, false);
        c = __builtin_amdgcn_wmma_f32_16x16x32_f16(false, a1, false, bf1, (short)0, c, false, false);

        // epilogue: lane holds branch n, pixels j = jslice + 8g + r
        float tv[8], feat[8];
        const int jb = jslice + 8 * g;
        if (n < NBR) {
            const float* tp = temporal + (size_t)n * NN + (size_t)i * NIMG + jb;
            float4 t0 = *(const float4*)tp;
            float4 t1 = *(const float4*)(tp + 4);
            tv[0]=t0.x; tv[1]=t0.y; tv[2]=t0.z; tv[3]=t0.w;
            tv[4]=t1.x; tv[5]=t1.y; tv[6]=t1.z; tv[7]=t1.w;
        } else {
#pragma unroll
            for (int r = 0; r < 8; ++r) tv[r] = 0.f;
        }
#pragma unroll
        for (int r = 0; r < 8; ++r) {
            feat[r] = (c[r] + bias) * tv[r];
            lmax = fmaxf(lmax, feat[r]);
        }
        // 3x3 non-overlapping average pool (cells straddle tiles -> atomicAdd)
        if (n < NBR && i < 510) {
            float* pbase = out + (size_t)bidx * OUTROW + NBR
                         + (size_t)n * POOLPER + (size_t)(i / 3) * POOLW;
            int r = 0;
            while (r < 8) {
                int j = jb + r;
                if (j >= 510) break;
                int pc = j / 3;
                float s = 0.f;
                while (r < 8 && (jb + r) < 510 && (jb + r) / 3 == pc) { s += feat[r]; ++r; }
                atomicAdd(pbase + pc, s * (1.0f / 9.0f));
            }
        }
    }

    // global max: lanes L and L+16 share branch n
    lmax = fmaxf(lmax, __shfl_xor(lmax, 16, 32));
    if (lane < 16) smax[wv][lane] = lmax;
    __syncthreads();
    if (tid < 16) {
        float m = smax[0][tid];
#pragma unroll
        for (int w = 1; w < 8; ++w) m = fmaxf(m, smax[w][tid]);
        if (tid < NBR)
            atomicMax((unsigned*)(out + (size_t)bidx * OUTROW + tid), enc_f32(m));
    }
}

// ---------------------------------------------------------------------------
// Decode the 16x14 encoded max slots back to float
// ---------------------------------------------------------------------------
__global__ void lip_finalize(float* __restrict__ out) {
    int idx = threadIdx.x;
    if (idx >= 16 * NBR) return;
    int b = idx / NBR, n = idx % NBR;
    unsigned* p = (unsigned*)(out + (size_t)b * OUTROW + n);
    unsigned u = *p;
    float f = (u & 0x80000000u) ? __uint_as_float(u & 0x7FFFFFFFu)
                                : __uint_as_float(~u);
    *p = __float_as_uint(f);
}

extern "C" void kernel_launch(void* const* d_in, const int* in_sizes, int n_in,
                              void* d_out, int out_size, void* d_ws, size_t ws_size,
                              hipStream_t stream) {
    (void)in_sizes; (void)ws_size;
    const float* x = (const float*)d_in[0];
    KPtrs kp;
    const float *biases, *gamma, *beta, *mmean, *mvar;
    if (n_in >= 20) {               // tuple expanded into 14 separate inputs
        for (int i = 0; i < NBR; ++i) kp.w[i] = (const float*)d_in[1 + i];
        biases = (const float*)d_in[15];
        gamma  = (const float*)d_in[16];
        beta   = (const float*)d_in[17];
        mmean  = (const float*)d_in[18];
        mvar   = (const float*)d_in[19];
    } else {                        // tuple flattened into one buffer
        const float* kb = (const float*)d_in[1];
        size_t off = 0;
        for (int i = 0; i < NBR; ++i) { kp.w[i] = kb + off; off += (size_t)h_KH[i] * h_KW[i]; }
        biases = (const float*)d_in[2];
        gamma  = (const float*)d_in[3];
        beta   = (const float*)d_in[4];
        mmean  = (const float*)d_in[5];
        mvar   = (const float*)d_in[6];
    }
    float*     tws = (float*)d_ws;
    _Float16*  Bw  = (_Float16*)((char*)d_ws + TEMP_BYTES);

    hipMemsetAsync(d_out, 0, (size_t)out_size * sizeof(float), stream);
    lip_prep<<<4, 256, 0, stream>>>(kp, Bw);
    lip_temporal<<<(NBR * NN + 255) / 256, 256, 0, stream>>>(tws);
    lip_main<<<dim3(NIMG / TILE_COLS, NIMG / TILE_ROWS, 16), 256, 0, stream>>>(
        x, biases, gamma, beta, mmean, mvar, tws, Bw, (float*)d_out);
    lip_finalize<<<1, 256, 0, stream>>>((float*)d_out);
}
